// Debiased_Representation_Loss_82154134438046
// MI455X (gfx1250) — compile-verified, matrix operand order checked by reading, not run on hardware
//
#include <hip/hip_runtime.h>

// ---------------------------------------------------------------------------
// Types for CDNA5 WMMA (wave32) + TDM descriptors
// ---------------------------------------------------------------------------
typedef __bf16 v16bf __attribute__((ext_vector_type(16)));
typedef __bf16 bf16x8 __attribute__((ext_vector_type(8)));
typedef float  v8f   __attribute__((ext_vector_type(8)));
typedef unsigned int u32x4 __attribute__((ext_vector_type(4)));
typedef int i32x4 __attribute__((ext_vector_type(4)));
typedef int i32x8 __attribute__((ext_vector_type(8)));

#define B_ROWS   4096
#define DIM      768
#define HID      256
#define NCLS     200
#define C_OLD    100
#define EPSILON_ 0.05f
#define TAU_INV  10.0f      // 1/0.1
#define EPS_     1e-8f
#define LOG2_    0.6931471805599453f
#define LOG100_  4.605170185988092f

// ---------------------------------------------------------------------------
// Workspace layout (bytes, all 256-aligned)
// ---------------------------------------------------------------------------
static constexpr size_t OFF_ACC  = 0;
static constexpr size_t OFF_PART = 1024;
static constexpr size_t PART_N   = (size_t)B_ROWS * 8;              // 32768
static constexpr size_t OFF_ZN   = OFF_PART + 5 * PART_N * 4;       // +655360
static constexpr size_t SZ_BD    = (size_t)B_ROWS * DIM * 2;        // 6291456
static constexpr size_t OFF_BN   = OFF_ZN + SZ_BD;
static constexpr size_t OFF_ZU   = OFF_BN + SZ_BD;
static constexpr size_t SZ_W     = (size_t)HID * DIM * 2;           // 393216
static constexpr size_t OFF_W1   = OFF_ZU + SZ_BD;
static constexpr size_t OFF_W2   = OFF_W1 + SZ_W;
static constexpr size_t SZ_F     = (size_t)B_ROWS * HID * 2;        // 2097152
static constexpr size_t OFF_F1   = OFF_W2 + SZ_W;
static constexpr size_t OFF_F2   = OFF_F1 + SZ_F;

// ---------------------------------------------------------------------------
// bf16 16x16x32 fragment loaders (ISA 7.12.2 layout):
// lanes 0-15 row M=lane, K={0..7,16..23}; lanes 16-31 row M=lane-16,
// K={8..15,24..31}.  Two 16-byte loads/lane -> global_load_b128/ds_load_b128.
// ---------------------------------------------------------------------------
__device__ __forceinline__ v16bf load_frag(const __bf16* __restrict__ base,
                                           int row0, int ld, int k0, int lane) {
  const int r = row0 + (lane & 15);
  const int k = k0 + ((lane & 16) ? 8 : 0);
  const __bf16* p = base + (size_t)r * ld + k;
  bf16x8 lo = *reinterpret_cast<const bf16x8*>(p);
  bf16x8 hi = *reinterpret_cast<const bf16x8*>(p + 16);
  v16bf f;
#pragma unroll
  for (int e = 0; e < 8; ++e) { f[e] = lo[e]; f[e + 8] = hi[e]; }
  return f;
}

__device__ __forceinline__ v16bf load_frag_lds(const __bf16* base, int ld,
                                               int k0, int lane) {
  const int r = lane & 15;                       // strip-local row
  const int k = k0 + ((lane & 16) ? 8 : 0);
  const __bf16* p = base + r * ld + k;
  bf16x8 lo = *reinterpret_cast<const bf16x8*>(p);
  bf16x8 hi = *reinterpret_cast<const bf16x8*>(p + 16);
  v16bf f;
#pragma unroll
  for (int e = 0; e < 8; ++e) { f[e] = lo[e]; f[e + 8] = hi[e]; }
  return f;
}

__device__ __forceinline__ v8f wmma_bf16(v16bf a, v16bf b, v8f c) {
  return __builtin_amdgcn_wmma_f32_16x16x32_bf16(
      false, a, false, b, (short)0, c, false, false);
}

// ---------------------------------------------------------------------------
// Tensor Data Mover: 2-D tile (rows x elems, bf16) global -> LDS.
// Descriptor per ISA 8.3/8.4: Group0 = {count=1, lds_addr, global_addr, type=2},
// Group1 = {data_size=2B, tensor_dim0/1, tile_dim0/1, tensor_dim0_stride}.
// Generic pointer to __shared__: low 32 bits of the flat address are the LDS
// byte offset (ISA 10.2 LDS aperture mapping).
// ---------------------------------------------------------------------------
__device__ __forceinline__ void tdm_load_2d(const void* gptr, uint32_t ldsOff,
                                            uint32_t tensor_d0, uint32_t tensor_d1,
                                            uint32_t tile_d0, uint32_t tile_d1,
                                            uint64_t stride0) {
  const uint64_t ga = (uint64_t)(uintptr_t)gptr;
  u32x4 g0;
  g0.x = 1u;                                                  // count=1, user mode
  g0.y = ldsOff;                                              // LDS byte address
  g0.z = (uint32_t)ga;                                        // global addr [31:0]
  g0.w = (uint32_t)((ga >> 32) & 0x01FFFFFFu) | (2u << 30);   // [56:32] + type=2
  i32x8 g1;
  g1[0] = (int)(1u << 16);                                    // data_size = 2 bytes
  g1[1] = (int)((tensor_d0 & 0xFFFFu) << 16);                 // td0[15:0]
  g1[2] = (int)(((tensor_d0 >> 16) & 0xFFFFu) |
                ((tensor_d1 & 0xFFFFu) << 16));               // td0[31:16], td1[15:0]
  g1[3] = (int)(((tensor_d1 >> 16) & 0xFFFFu) |
                ((tile_d0 & 0xFFFFu) << 16));                 // td1[31:16], tile0
  g1[4] = (int)(tile_d1 & 0xFFFFu);                           // tile1 (tile2=0)
  g1[5] = (int)(stride0 & 0xFFFFFFFFu);                       // dim0 stride [31:0]
  g1[6] = (int)((stride0 >> 32) & 0xFFFFu);                   // dim0 stride [47:32]
  g1[7] = 0;
  i32x4 gz = {0, 0, 0, 0};
#if defined(__clang_major__) && (__clang_major__ >= 23)
  i32x8 gz8 = {0, 0, 0, 0, 0, 0, 0, 0};
  __builtin_amdgcn_tensor_load_to_lds(g0, g1, gz, gz, gz8, 0);
#else
  __builtin_amdgcn_tensor_load_to_lds(g0, g1, gz, gz, 0);
#endif
}

// ---------------------------------------------------------------------------
// Kernel 1: softmax over active classes, accumulate mean probs.
// ---------------------------------------------------------------------------
__global__ __launch_bounds__(256) void prob_kernel(
    const float* __restrict__ logits, const int* __restrict__ oldIdx,
    const int* __restrict__ newIdx, float* __restrict__ acc) {
  __shared__ float sAcc[NCLS];
  const int tid = threadIdx.x;
  if (tid < NCLS) sAcc[tid] = 0.0f;
  __syncthreads();

  const int lane = tid & 31;
  const int row  = blockIdx.x * 8 + (tid >> 5);

  float v[7];
  float mx = -__builtin_inff();
#pragma unroll
  for (int t = 0; t < 7; ++t) {
    const int c = lane + t * 32;
    float x = -__builtin_inff();
    if (c < NCLS) {
      const int col = (c < C_OLD) ? oldIdx[c] : newIdx[c - C_OLD];
      x = logits[(size_t)row * NCLS + col];
    }
    v[t] = x; mx = fmaxf(mx, x);
  }
#pragma unroll
  for (int off = 1; off < 32; off <<= 1) mx = fmaxf(mx, __shfl_xor(mx, off, 32));
  float se = 0.0f;
#pragma unroll
  for (int t = 0; t < 7; ++t) {
    const float e = (lane + t * 32 < NCLS) ? __expf(v[t] - mx) : 0.0f;
    v[t] = e; se += e;
  }
#pragma unroll
  for (int off = 1; off < 32; off <<= 1) se += __shfl_xor(se, off, 32);
  const float inv = 1.0f / se;
#pragma unroll
  for (int t = 0; t < 7; ++t) {
    const int c = lane + t * 32;
    if (c < NCLS) atomicAdd(&sAcc[c], v[t] * inv);
  }
  __syncthreads();
  if (tid < NCLS) atomicAdd(&acc[tid], sAcc[tid] * (1.0f / (float)B_ROWS));
}

// ---------------------------------------------------------------------------
// Kernel 2: row L2-normalize z_u and base_features; emit bf16 Zn, Bn, Zu.
// ---------------------------------------------------------------------------
__global__ __launch_bounds__(256) void norm_kernel(
    const float* __restrict__ z, const float* __restrict__ bf,
    __bf16* __restrict__ Zn, __bf16* __restrict__ Bn, __bf16* __restrict__ Zu) {
  __shared__ float red[256];
  const int row = blockIdx.x, tid = threadIdx.x;
  const size_t rb = (size_t)row * DIM;

  float zz[3]; float s = 0.0f;
#pragma unroll
  for (int e = 0; e < 3; ++e) { zz[e] = z[rb + tid + e * 256]; s += zz[e] * zz[e]; }
  red[tid] = s; __syncthreads();
  for (int st = 128; st > 0; st >>= 1) { if (tid < st) red[tid] += red[tid + st]; __syncthreads(); }
  const float scz = 1.0f / fmaxf(sqrtf(red[0]), 1e-12f);
  __syncthreads();
#pragma unroll
  for (int e = 0; e < 3; ++e) {
    Zn[rb + tid + e * 256] = (__bf16)(zz[e] * scz);
    Zu[rb + tid + e * 256] = (__bf16)zz[e];
  }

  float bb[3]; s = 0.0f;
#pragma unroll
  for (int e = 0; e < 3; ++e) { bb[e] = bf[rb + tid + e * 256]; s += bb[e] * bb[e]; }
  red[tid] = s; __syncthreads();
  for (int st = 128; st > 0; st >>= 1) { if (tid < st) red[tid] += red[tid + st]; __syncthreads(); }
  const float scb = 1.0f / fmaxf(sqrtf(red[0]), 1e-12f);
  __syncthreads();
#pragma unroll
  for (int e = 0; e < 3; ++e) Bn[rb + tid + e * 256] = (__bf16)(bb[e] * scb);
}

// ---------------------------------------------------------------------------
// Kernel 3: convert FC weights to bf16.
// ---------------------------------------------------------------------------
__global__ __launch_bounds__(256) void cvt_kernel(
    const float* __restrict__ w1, const float* __restrict__ w2,
    __bf16* __restrict__ W1, __bf16* __restrict__ W2) {
  const int i = blockIdx.x * 256 + threadIdx.x;
  if (i < HID * DIM) { W1[i] = (__bf16)w1[i]; W2[i] = (__bf16)w2[i]; }
}

// ---------------------------------------------------------------------------
// Kernel 4: FC GEMMs  F{1,2}z = z_u @ W{1,2}ᵀ + b  (4096x256, K=768), WMMA.
// ---------------------------------------------------------------------------
__global__ __launch_bounds__(256) void fc_kernel(
    const __bf16* __restrict__ Zu, const __bf16* __restrict__ W1,
    const __bf16* __restrict__ W2, const float* __restrict__ b1,
    const float* __restrict__ b2, __bf16* __restrict__ F1,
    __bf16* __restrict__ F2) {
  const int lane = threadIdx.x & 31;
  const int tile = blockIdx.x * 8 + (threadIdx.x >> 5);   // 4096 tiles
  const int m0 = (tile >> 4) * 16;
  const int n0 = (tile & 15) * 16;
  v8f c1 = {}; v8f c2 = {};
  for (int kc = 0; kc < DIM; kc += 32) {
    v16bf a  = load_frag(Zu, m0, DIM, kc, lane);
    v16bf w1 = load_frag(W1, n0, DIM, kc, lane);
    c1 = wmma_bf16(a, w1, c1);
    v16bf w2 = load_frag(W2, n0, DIM, kc, lane);
    c2 = wmma_bf16(a, w2, c2);
  }
  const int gn = n0 + (lane & 15);
  const float bias1 = b1[gn], bias2 = b2[gn];
#pragma unroll
  for (int p = 0; p < 8; ++p) {
    const int gm = m0 + p + ((lane & 16) ? 8 : 0);
    F1[(size_t)gm * HID + gn] = (__bf16)(c1[p] + bias1);
    F2[(size_t)gm * HID + gn] = (__bf16)(c2[p] + bias2);
  }
}

// ---------------------------------------------------------------------------
// Kernel 5: fused flash-style contrastive pass.
// Workgroup = one 16-row strip; TDM stages the strip's A-side operands
// (Zn, Bn, F1z rows) into LDS once; 8 waves split the 4096 columns into 8
// chunks of 512.  A fragments come from LDS (ds_load_b128), B fragments
// stream from L2 (global_load_b128).  Online (m, Z, S, denom, nn) per row.
// ---------------------------------------------------------------------------
__global__ __launch_bounds__(256) void fused_kernel(
    const __bf16* __restrict__ Zn, const __bf16* __restrict__ Bn,
    const __bf16* __restrict__ F1, const __bf16* __restrict__ F2,
    float* __restrict__ Pm, float* __restrict__ Pz, float* __restrict__ Ps,
    float* __restrict__ Pd, float* __restrict__ Pn) {
  __shared__ __align__(16) __bf16 sZn[16 * DIM];
  __shared__ __align__(16) __bf16 sBn[16 * DIM];
  __shared__ __align__(16) __bf16 sF1[16 * HID];

  const int lane  = threadIdx.x & 31;
  const int wave  = threadIdx.x >> 5;
  const int i0    = blockIdx.x * 16;         // 256 strips
  const int chunk = wave;                    // 8 chunks of 512 columns
  const int jBeg  = chunk * 512;

  if (wave == 0) {   // TDM ops are wave-level (EXEC ignored); issue from wave 0
    tdm_load_2d(Zn + (size_t)i0 * DIM, (uint32_t)(uintptr_t)&sZn[0],
                DIM, B_ROWS, DIM, 16, DIM);
    tdm_load_2d(Bn + (size_t)i0 * DIM, (uint32_t)(uintptr_t)&sBn[0],
                DIM, B_ROWS, DIM, 16, DIM);
    tdm_load_2d(F1 + (size_t)i0 * HID, (uint32_t)(uintptr_t)&sF1[0],
                HID, B_ROWS, HID, 16, HID);
    __builtin_amdgcn_s_wait_tensorcnt(0);
  }
  __syncthreads();

  float m[8], Z[8], S[8], dn[8], nn[8];
#pragma unroll
  for (int p = 0; p < 8; ++p) {
    m[p] = -__builtin_inff(); Z[p] = 0.f; S[p] = 0.f; dn[p] = 0.f; nn[p] = 0.f;
  }

  for (int jt = 0; jt < 32; ++jt) {
    const int j0 = jBeg + jt * 16;
    if (jt + 1 < 32)
      __builtin_prefetch(Zn + (size_t)(j0 + 16 + (lane & 15)) * DIM, 0, 1);

    v8f c_s = {}; v8f c_b = {}; v8f c_a = {};
    for (int kc = 0; kc < DIM; kc += 32) {
      v16bf a1 = load_frag_lds(sZn, DIM, kc, lane);
      v16bf b1 = load_frag(Zn, j0, DIM, kc, lane);
      c_s = wmma_bf16(a1, b1, c_s);
      v16bf a2 = load_frag_lds(sBn, DIM, kc, lane);
      v16bf b2 = load_frag(Bn, j0, DIM, kc, lane);
      c_b = wmma_bf16(a2, b2, c_b);
    }
#pragma unroll
    for (int kc = 0; kc < HID; kc += 32) {
      v16bf a3 = load_frag_lds(sF1, HID, kc, lane);
      v16bf b3 = load_frag(F2, j0, HID, kc, lane);
      c_a = wmma_bf16(a3, b3, c_a);
    }

    const int cj = j0 + (lane & 15);
    const int rOff = (lane & 16) ? 8 : 0;
#pragma unroll
    for (int p = 0; p < 8; ++p) {
      const int ri = i0 + p + rOff;
      const float bs = c_b[p];
      const float sv = c_s[p] * TAU_INV;
      const float av = c_a[p];
      const bool diag = (ri == cj);
      dn[p] += diag ? 0.0f : __expf(sv);
      if ((bs > EPSILON_) && !diag) {
        nn[p] += 1.0f;
        const float mo = m[p];
        const float mn = fmaxf(mo, av);
        const float sc = (mo == -__builtin_inff()) ? 0.0f : __expf(mo - mn);
        const float w  = __expf(av - mn);
        Z[p] = Z[p] * sc + w;
        S[p] = S[p] * sc + w * sv;
        m[p] = mn;
      }
    }
  }

  // Cross-lane combine within each 16-lane half (one half per 8 rows).
#pragma unroll
  for (int p = 0; p < 8; ++p) {
    float mm = m[p], ZZ = Z[p], SS = S[p], dd = dn[p], cc = nn[p];
#pragma unroll
    for (int off = 1; off < 16; off <<= 1) {
      const float mo = __shfl_xor(mm, off, 16);
      const float Zo = __shfl_xor(ZZ, off, 16);
      const float So = __shfl_xor(SS, off, 16);
      dd += __shfl_xor(dd, off, 16);
      cc += __shfl_xor(cc, off, 16);
      const float mn = fmaxf(mm, mo);
      const float s1 = (mm == -__builtin_inff()) ? 0.0f : __expf(mm - mn);
      const float s2 = (mo == -__builtin_inff()) ? 0.0f : __expf(mo - mn);
      ZZ = ZZ * s1 + Zo * s2;
      SS = SS * s1 + So * s2;
      mm = mn;
    }
    if ((lane & 15) == 0) {
      const int row = i0 + p + ((lane & 16) ? 8 : 0);
      const int idx = row * 8 + chunk;
      Pm[idx] = mm; Pz[idx] = ZZ; Ps[idx] = SS; Pd[idx] = dd; Pn[idx] = cc;
    }
  }
}

// ---------------------------------------------------------------------------
// Kernel 6: merge 8 chunk-partials per row, per-sample loss, block reduce.
// ---------------------------------------------------------------------------
__global__ __launch_bounds__(256) void merge_kernel(
    const float* __restrict__ Pm, const float* __restrict__ Pz,
    const float* __restrict__ Ps, const float* __restrict__ Pd,
    const float* __restrict__ Pn, float* __restrict__ acc) {
  const int tid = threadIdx.x;
  const int row = blockIdx.x * 256 + tid;   // grid 16 -> exactly 4096 rows
  float ps = 0.0f, pv = 0.0f;
  {
    float mm = -__builtin_inff(), ZZ = 0.f, SS = 0.f, dd = 0.f, cc = 0.f;
#pragma unroll
    for (int c = 0; c < 8; ++c) {
      const int idx = row * 8 + c;
      const float mo = Pm[idx], Zo = Pz[idx], So = Ps[idx];
      dd += Pd[idx]; cc += Pn[idx];
      const float mn = fmaxf(mm, mo);
      const float s1 = (mm == -__builtin_inff()) ? 0.0f : __expf(mm - mn);
      const float s2 = (mo == -__builtin_inff()) ? 0.0f : __expf(mo - mn);
      ZZ = ZZ * s1 + Zo * s2;
      SS = SS * s1 + So * s2;
      mm = mn;
    }
    if (cc > 0.0f) {
      const float loss = __logf(dd + EPS_) - SS / ZZ;
      ps = loss / cc; pv = 1.0f;
    }
  }
  __shared__ float r1[256], r2[256];
  r1[tid] = ps; r2[tid] = pv; __syncthreads();
  for (int st = 128; st > 0; st >>= 1) {
    if (tid < st) { r1[tid] += r1[tid + st]; r2[tid] += r2[tid + st]; }
    __syncthreads();
  }
  if (tid == 0) { atomicAdd(&acc[200], r1[0]); atomicAdd(&acc[201], r2[0]); }
}

// ---------------------------------------------------------------------------
// Kernel 7: entropy terms + final scalar.
// ---------------------------------------------------------------------------
__global__ void finalize_kernel(const float* __restrict__ acc,
                                float* __restrict__ out) {
  if (threadIdx.x == 0) {
    float pold = 0.f, pnew = 0.f;
    for (int c = 0; c < C_OLD; ++c) pold += acc[c];
    for (int c = C_OLD; c < NCLS; ++c) pnew += acc[c];
    float li = pold * __logf(pold + EPS_) + pnew * __logf(pnew + EPS_) + LOG2_;
    float lo = 0.f, ln = 0.f;
    for (int c = 0; c < C_OLD; ++c) {
      const float q = acc[c] / (pold + EPS_);
      lo += q * __logf(q + EPS_);
    }
    lo += LOG100_;
    for (int c = C_OLD; c < NCLS; ++c) {
      const float q = acc[c] / (pnew + EPS_);
      ln += q * __logf(q + EPS_);
    }
    ln += LOG100_;
    const float cnt = acc[201];
    const float contr = (cnt > 0.f) ? acc[200] / cnt : 0.f;
    out[0] = li + lo + ln + contr;
  }
}

// ---------------------------------------------------------------------------
extern "C" void kernel_launch(void* const* d_in, const int* in_sizes, int n_in,
                              void* d_out, int out_size, void* d_ws,
                              size_t ws_size, hipStream_t stream) {
  (void)in_sizes; (void)n_in; (void)out_size; (void)ws_size;
  const float* z_u    = (const float*)d_in[0];
  const float* logits = (const float*)d_in[1];
  const int*   oldIdx = (const int*)d_in[2];
  const int*   newIdx = (const int*)d_in[3];
  const float* basef  = (const float*)d_in[4];
  const float* f1w    = (const float*)d_in[5];
  const float* f1b    = (const float*)d_in[6];
  const float* f2w    = (const float*)d_in[7];
  const float* f2b    = (const float*)d_in[8];

  char* ws = (char*)d_ws;
  float* acc = (float*)(ws + OFF_ACC);
  float* Pm  = (float*)(ws + OFF_PART);
  float* Pz  = Pm + PART_N;
  float* Ps  = Pz + PART_N;
  float* Pd  = Ps + PART_N;
  float* Pn  = Pd + PART_N;
  __bf16* Zn = (__bf16*)(ws + OFF_ZN);
  __bf16* Bn = (__bf16*)(ws + OFF_BN);
  __bf16* Zu = (__bf16*)(ws + OFF_ZU);
  __bf16* W1 = (__bf16*)(ws + OFF_W1);
  __bf16* W2 = (__bf16*)(ws + OFF_W2);
  __bf16* F1 = (__bf16*)(ws + OFF_F1);
  __bf16* F2 = (__bf16*)(ws + OFF_F2);

  hipMemsetAsync(acc, 0, 1024, stream);
  prob_kernel<<<512, 256, 0, stream>>>(logits, oldIdx, newIdx, acc);
  norm_kernel<<<B_ROWS, 256, 0, stream>>>(z_u, basef, Zn, Bn, Zu);
  cvt_kernel<<<(HID * DIM + 255) / 256, 256, 0, stream>>>(f1w, f2w, W1, W2);
  fc_kernel<<<512, 256, 0, stream>>>(Zu, W1, W2, f1b, f2b, F1, F2);
  fused_kernel<<<256, 256, 0, stream>>>(Zn, Bn, F1, F2, Pm, Pz, Ps, Pd, Pn);
  merge_kernel<<<16, 256, 0, stream>>>(Pm, Pz, Ps, Pd, Pn, acc);
  finalize_kernel<<<1, 32, 0, stream>>>(acc, (float*)d_out);
}
